// TransformerBlock_6425271075386
// MI455X (gfx1250) — compile-verified, matrix operand order checked by reading, not run on hardware
//
#include <hip/hip_runtime.h>
#include <hip/hip_bf16.h>

typedef __attribute__((ext_vector_type(16))) _Float16 v16h;
typedef __attribute__((ext_vector_type(8)))  _Float16 v8h;
typedef __attribute__((ext_vector_type(8)))  float    v8f;
typedef __attribute__((ext_vector_type(4)))  float    v4f;
typedef __attribute__((ext_vector_type(4)))  unsigned int v4u;
typedef __attribute__((ext_vector_type(8)))  int      v8i;
typedef __attribute__((ext_vector_type(4)))  int      v4i;

constexpr int   kT      = 1024;
constexpr int   kHid    = 2880;
constexpr int   kNH     = 64;
constexpr int   kKV     = 8;
constexpr int   kHD     = 64;
constexpr int   kInter  = 2880;
constexpr int   kQkvD   = kHD * (kNH + 2 * kKV);   // 5120
constexpr float kSmScale = 0.125f;                 // 1/sqrt(64)
constexpr float kAlpha   = 1.702f;
constexpr float kLimit   = 7.0f;
constexpr float kEps     = 1e-5f;
constexpr float kRopeBase = 150000.0f;
constexpr float kInitCtx  = 4096.0f;
constexpr float kScaling  = 32.0f;
constexpr float kNtkAlpha = 1.0f;
constexpr float kNtkBeta  = 32.0f;

// ---------------------------------------------------------------------------
// WMMA helpers (CDNA5 layouts, ISA 7.12.2).
//   16-bit A-frag 16x32: lane=(m=lane%16, h=lane/16); elems 0..7  -> K = h*8..h*8+7
//                                                     elems 8..15 -> K = 16+h*8..+7
//   16-bit B-frag 32x16: same pattern with n=lane%16 indexing rows of W[N][K].
//   f32 C/D 16x16:       lane holds (m = r + 8*h, n = lane%16) in VGPR r.
// ---------------------------------------------------------------------------
__device__ __forceinline__ v8f wmma_f16(v16h a, v16h b, v8f c) {
  return __builtin_amdgcn_wmma_f32_16x16x32_f16(false, a, false, b, (short)0, c,
                                                false, false);
}

// f16 fragment from row-major f16 tile, leading dim ld (halfs).
__device__ __forceinline__ v16h ld_frag(const _Float16* __restrict__ base, int ld) {
  const int lane = threadIdx.x & 31;
  const int r = lane & 15;
  const int h = lane >> 4;
  const _Float16* rowp = base + r * ld;
  v8h lo = *(const v8h*)(rowp + h * 8);
  v8h hi = *(const v8h*)(rowp + 16 + h * 8);
  v16h f;
#pragma unroll
  for (int i = 0; i < 8; ++i) { f[i] = lo[i]; f[i + 8] = hi[i]; }
  return f;
}

// f16 fragment from row-major f32 tile (convert at load), leading dim ld (floats).
__device__ __forceinline__ v16h ld_frag_f32(const float* __restrict__ base, int ld) {
  const int lane = threadIdx.x & 31;
  const int r = lane & 15;
  const int h = lane >> 4;
  const float* rowp = base + r * ld;
  v4f a0 = *(const v4f*)(rowp + h * 8);
  v4f a1 = *(const v4f*)(rowp + h * 8 + 4);
  v4f b0 = *(const v4f*)(rowp + 16 + h * 8);
  v4f b1 = *(const v4f*)(rowp + 16 + h * 8 + 4);
  v16h f;
#pragma unroll
  for (int i = 0; i < 4; ++i) {
    f[i]      = (_Float16)a0[i];
    f[4 + i]  = (_Float16)a1[i];
    f[8 + i]  = (_Float16)b0[i];
    f[12 + i] = (_Float16)b1[i];
  }
  return f;
}

// ---------------------------------------------------------------------------
// Tensor Data Mover: issue a 2D tile load (ISA ch.8 D# layout).
//   g0: count=1 | lds_addr | global_addr[31:0] | global_addr[56:32] + type=2
//   g1: data_size | tensor_dim0/1 | tile_dim0/1 | tensor_dim0_stride
// Groups 2/3 zero (<=2D tensor). Tracked by TENSORcnt.
// ---------------------------------------------------------------------------
__device__ __forceinline__ void tdm_load_2d(unsigned lds_addr, const void* gptr,
                                            unsigned data_size_code,  // 1=2B, 2=4B
                                            unsigned tile_d0, unsigned tile_d1,
                                            unsigned tensor_d0, unsigned tensor_d1,
                                            unsigned long stride0) {
  const unsigned long ga = (unsigned long)gptr;
  v4u g0;
  g0[0] = 1u;                                         // count=1, user mode
  g0[1] = lds_addr;                                   // LDS byte address
  g0[2] = (unsigned)(ga & 0xffffffffu);               // global addr [31:0]
  g0[3] = (unsigned)((ga >> 32) & 0x01ffffffu)        // global addr [56:32]
          | (2u << 30);                               // type = 2 ("image")
  v8i g1;
  g1[0] = (int)(data_size_code << 16);                // workgroup_mask=0, data_size
  g1[1] = (int)((tensor_d0 & 0xffffu) << 16);         // td0[15:0] -> bits[63:48]
  g1[2] = (int)(((tensor_d0 >> 16) & 0xffffu) |       // td0[31:16]
                ((tensor_d1 & 0xffffu) << 16));       // td1[15:0]
  g1[3] = (int)(((tensor_d1 >> 16) & 0xffffu) |       // td1[31:16]
                ((tile_d0 & 0xffffu) << 16));         // tile_dim0
  g1[4] = (int)(tile_d1 & 0xffffu);                   // tile_dim1 (tile_dim2=0)
  g1[5] = (int)(stride0 & 0xffffffffu);               // stride0 [31:0]
  g1[6] = (int)((stride0 >> 32) & 0xffffu);           // stride0 [47:32], stride1_lo=0
  g1[7] = 0;
  const v4i z4 = {0, 0, 0, 0};
#if __clang_major__ >= 23
  const v8i z8 = {0, 0, 0, 0, 0, 0, 0, 0};
  __builtin_amdgcn_tensor_load_to_lds(g0, g1, z4, z4, z8, 0);
#else
  __builtin_amdgcn_tensor_load_to_lds(g0, g1, z4, z4, 0);
#endif
}

// ---------------------------------------------------------------------------
// Kernel 1: RMSNorm -> f16
// ---------------------------------------------------------------------------
__global__ __launch_bounds__(256) void rmsnorm_h(const float* __restrict__ x,
                                                 const float* __restrict__ scale,
                                                 _Float16* __restrict__ out) {
  __shared__ float red[8];
  __shared__ float s_inv;
  const int row = blockIdx.x;
  const float* xr = x + (long)row * kHid;
  float ss = 0.f;
  for (int i = threadIdx.x; i < kHid; i += 256) { float v = xr[i]; ss += v * v; }
#pragma unroll
  for (int off = 16; off > 0; off >>= 1) ss += __shfl_xor(ss, off, 32);
  if ((threadIdx.x & 31) == 0) red[threadIdx.x >> 5] = ss;
  __syncthreads();
  if (threadIdx.x == 0) {
    float tot = 0.f;
#pragma unroll
    for (int i = 0; i < 8; ++i) tot += red[i];
    s_inv = rsqrtf(tot / (float)kHid + kEps);
  }
  __syncthreads();
  const float inv = s_inv;
  for (int i = threadIdx.x; i < kHid; i += 256)
    out[(long)row * kHid + i] = (_Float16)(xr[i] * inv * scale[i]);
}

// ---------------------------------------------------------------------------
// Kernel 2: GEMM  C[M,N] = A_h[M,K] * W[N,K]^T + bias (+ residual)
// 256 thr (8 waves), tile BM=128 x BN=64, K-step 32.
// TDM double-buffered LDS staging: wave 0 DMAs tile k+1 while all waves run
// WMMA on tile k; sync = s_wait_tensorcnt + workgroup barrier.
// ---------------------------------------------------------------------------
constexpr int kBM = 128;
constexpr int kBN = 64;
constexpr int kBK = 32;

__global__ __launch_bounds__(256) void gemm_h(const _Float16* __restrict__ A,
                                              const float* __restrict__ W,
                                              const float* __restrict__ bias,
                                              const float* __restrict__ resid,
                                              float* __restrict__ C,
                                              int M, int N, int K) {
  __shared__ _Float16 sA[2][kBM * kBK];   // 2 x 8KB, rows of 32 halfs (64B)
  __shared__ float    sB[2][kBN * kBK];   // 2 x 8KB, rows of 32 floats (128B)
  const int tid = threadIdx.x;
  const int wave = tid >> 5;
  const int wm = wave >> 1;               // 0..3 -> M sub-tile of 32
  const int wn = wave & 1;                // 0..1 -> N sub-tile of 32
  const int bm = blockIdx.y * kBM;
  const int bn = blockIdx.x * kBN;

  v8f acc[2][2] = {};
  const int ktiles = K / kBK;

  if (wave == 0) {  // prologue DMA: tile 0
    tdm_load_2d((unsigned)(uintptr_t)&sA[0][0], &A[(long)bm * K],
                1, kBK, kBM, (unsigned)K, (unsigned)M, (unsigned long)K);
    tdm_load_2d((unsigned)(uintptr_t)&sB[0][0], &W[(long)bn * K],
                2, kBK, kBN, (unsigned)K, (unsigned)N, (unsigned long)K);
  }

  for (int kt = 0; kt < ktiles; ++kt) {
    if (wave == 0) __builtin_amdgcn_s_wait_tensorcnt(0);
    __syncthreads();                      // tile kt ready; buffer kt+1 free
    if (wave == 0 && kt + 1 < ktiles) {
      const long k1 = (long)(kt + 1) * kBK;
      tdm_load_2d((unsigned)(uintptr_t)&sA[(kt + 1) & 1][0], &A[(long)bm * K + k1],
                  1, kBK, kBM, (unsigned)K, (unsigned)M, (unsigned long)K);
      tdm_load_2d((unsigned)(uintptr_t)&sB[(kt + 1) & 1][0], &W[(long)bn * K + k1],
                  2, kBK, kBN, (unsigned)K, (unsigned)N, (unsigned long)K);
    }
    const _Float16* a = &sA[kt & 1][0];
    const float*    b = &sB[kt & 1][0];
    v16h bf0 = ld_frag_f32(b + (wn * 32 + 0) * kBK, kBK);
    v16h bf1 = ld_frag_f32(b + (wn * 32 + 16) * kBK, kBK);
    v16h af0 = ld_frag(a + (wm * 32 + 0) * kBK, kBK);
    acc[0][0] = wmma_f16(af0, bf0, acc[0][0]);
    acc[0][1] = wmma_f16(af0, bf1, acc[0][1]);
    v16h af1 = ld_frag(a + (wm * 32 + 16) * kBK, kBK);
    acc[1][0] = wmma_f16(af1, bf0, acc[1][0]);
    acc[1][1] = wmma_f16(af1, bf1, acc[1][1]);
  }

  const int lane = tid & 31;
  const int ln = lane & 15;
  const int lh = lane >> 4;
#pragma unroll
  for (int i = 0; i < 2; ++i) {
#pragma unroll
    for (int j = 0; j < 2; ++j) {
      const int gn = bn + wn * 32 + j * 16 + ln;
      const float b = bias ? bias[gn] : 0.f;
#pragma unroll
      for (int r = 0; r < 8; ++r) {
        const int gm = bm + wm * 32 + i * 16 + lh * 8 + r;
        float v = acc[i][j][r] + b;
        if (resid) v += resid[(long)gm * N + gn];
        C[(long)gm * N + gn] = v;
      }
    }
  }
}

// ---------------------------------------------------------------------------
// Kernel 3: RoPE (YaRN-style) + scatter to head-major Q/K (f16) and V^T (f16)
// qkv: [T][5120] f32. Qh:[NH][T][HD]  Kh:[KV][T][HD]  Vt:[KV][HD][T]
// ---------------------------------------------------------------------------
__global__ void rope_scatter(const float* __restrict__ qkv,
                             _Float16* __restrict__ Qh,
                             _Float16* __restrict__ Kh,
                             _Float16* __restrict__ Vt) {
  const int idx = blockIdx.x * blockDim.x + threadIdx.x;
  const int total = kT * (kNH + 2 * kKV) * 32;
  if (idx >= total) return;
  const int t = idx / ((kNH + 2 * kKV) * 32);
  const int rem = idx % ((kNH + 2 * kKV) * 32);
  const int hh = rem >> 5;
  const int d = rem & 31;
  const float* row = qkv + (long)t * kQkvD + hh * kHD;

  if (hh < kNH + kKV) {
    const float x1 = row[d];
    const float x2 = row[d + 32];
    const float freq = __powf(kRopeBase, (2.0f * (float)d) / (float)kHD);
    const float conc = 0.1f * __logf(kScaling) + 1.0f;
    const float dh = (float)kHD * 0.5f;
    const float twoPi = 6.28318530717958647692f;
    const float low  = dh * __logf(kInitCtx / (kNtkBeta  * twoPi)) / __logf(kRopeBase);
    const float high = dh * __logf(kInitCtx / (kNtkAlpha * twoPi)) / __logf(kRopeBase);
    const float interp = 1.0f / (kScaling * freq);
    const float extrap = 1.0f / freq;
    const float ramp = ((float)d - low) / (high - low);
    const float mask = 1.0f - fminf(fmaxf(ramp, 0.f), 1.f);
    const float invf = interp * (1.0f - mask) + extrap * mask;
    const float ang = (float)t * invf;
    const float c = __cosf(ang) * conc;
    const float s = __sinf(ang) * conc;
    const float o1 = x1 * c - x2 * s;
    const float o2 = x2 * c + x1 * s;
    if (hh < kNH) {
      _Float16* dst = Qh + ((long)hh * kT + t) * kHD;
      dst[d] = (_Float16)o1;
      dst[d + 32] = (_Float16)o2;
    } else {
      _Float16* dst = Kh + ((long)(hh - kNH) * kT + t) * kHD;
      dst[d] = (_Float16)o1;
      dst[d + 32] = (_Float16)o2;
    }
  } else {
    const int vh = hh - kNH - kKV;
    Vt[((long)vh * kHD + d) * kT + t] = (_Float16)row[d];
    Vt[((long)vh * kHD + d + 32) * kT + t] = (_Float16)row[d + 32];
  }
}

// ---------------------------------------------------------------------------
// Kernel 4: causal flash attention, 1 wave per (head, 16-query tile).
// Computes S^T = K * Q^T so softmax stats are per-lane (+1 xor16 shuffle) and
// P^T registers are directly the B-fragment for O^T = V^T * P^T.
// ---------------------------------------------------------------------------
__global__ __launch_bounds__(32) void attn_kernel(const _Float16* __restrict__ Qh,
                                                  const _Float16* __restrict__ Kh,
                                                  const _Float16* __restrict__ Vt,
                                                  _Float16* __restrict__ Oh) {
  const int head = blockIdx.x;
  const int qt = blockIdx.y;
  const int kvh = head >> 3;              // QMULT = 8
  const int lane = threadIdx.x;
  const int ln = lane & 15;
  const int lh = lane >> 4;

  const _Float16* Qbase = Qh + ((long)head * kT + qt * 16) * kHD;
  const _Float16* Kbase = Kh + (long)kvh * kT * kHD;
  const _Float16* Vbase = Vt + (long)kvh * kHD * kT;

  const v16h bq0 = ld_frag(Qbase + 0, kHD);
  const v16h bq1 = ld_frag(Qbase + 32, kHD);

  v8f acc[4] = {};
  float m_run = -1e30f, l_run = 0.f;
  const int q = qt * 16 + ln;
  const int nchunk = (qt * 16 + 16 + 31) / 32;

  for (int c = 0; c < nchunk; ++c) {
    const int kb = c * 32;
    v8f s0 = {}, s1 = {};
    v16h ak;
    ak = ld_frag(Kbase + (long)(kb + 0) * kHD + 0, kHD);   s0 = wmma_f16(ak, bq0, s0);
    ak = ld_frag(Kbase + (long)(kb + 0) * kHD + 32, kHD);  s0 = wmma_f16(ak, bq1, s0);
    ak = ld_frag(Kbase + (long)(kb + 16) * kHD + 0, kHD);  s1 = wmma_f16(ak, bq0, s1);
    ak = ld_frag(Kbase + (long)(kb + 16) * kHD + 32, kHD); s1 = wmma_f16(ak, bq1, s1);

    float cmax = -1e30f;
#pragma unroll
    for (int r = 0; r < 8; ++r) {
      const int k0i = kb + lh * 8 + r;
      const int k1i = kb + 16 + lh * 8 + r;
      s0[r] = (k0i <= q) ? s0[r] * kSmScale : -1e30f;
      s1[r] = (k1i <= q) ? s1[r] * kSmScale : -1e30f;
      cmax = fmaxf(cmax, fmaxf(s0[r], s1[r]));
    }
    cmax = fmaxf(cmax, __shfl_xor(cmax, 16, 32));
    const float m_new = fmaxf(m_run, cmax);
    const float corr = __expf(m_run - m_new);
    float lsum = 0.f;
    v16h pfrag;
#pragma unroll
    for (int r = 0; r < 8; ++r) {
      const float p0 = __expf(s0[r] - m_new);
      const float p1 = __expf(s1[r] - m_new);
      lsum += p0 + p1;
      pfrag[r] = (_Float16)p0;
      pfrag[8 + r] = (_Float16)p1;
    }
    lsum += __shfl_xor(lsum, 16, 32);
    l_run = l_run * corr + lsum;
    m_run = m_new;

#pragma unroll
    for (int mt = 0; mt < 4; ++mt) {
#pragma unroll
      for (int r = 0; r < 8; ++r) acc[mt][r] *= corr;
      const v16h av = ld_frag(Vbase + (long)(mt * 16) * kT + kb, kT);
      acc[mt] = wmma_f16(av, pfrag, acc[mt]);
    }
  }

  const float inv_l = 1.0f / l_run;
  const long row = (long)(qt * 16 + ln) * (kNH * kHD) + head * kHD;
#pragma unroll
  for (int mt = 0; mt < 4; ++mt) {
    v8h ov;
#pragma unroll
    for (int r = 0; r < 8; ++r) ov[r] = (_Float16)(acc[mt][r] * inv_l);
    *(v8h*)(&Oh[row + mt * 16 + lh * 8]) = ov;
  }
}

// ---------------------------------------------------------------------------
// Kernel 5: clamped SwiGLU: act[t,i] = g*sigmoid(1.702 g)*(lin+1), f16 out
// ---------------------------------------------------------------------------
__global__ void swiglu_act(const float* __restrict__ u, _Float16* __restrict__ act) {
  const int idx = blockIdx.x * blockDim.x + threadIdx.x;
  if (idx >= kT * kInter) return;
  const int t = idx / kInter;
  const int i = idx % kInter;
  const float* ur = u + (long)t * 2 * kInter;
  const float g = fminf(ur[2 * i], kLimit);
  const float lin = fminf(fmaxf(ur[2 * i + 1], -kLimit), kLimit);
  const float sg = 1.0f / (1.0f + __expf(-kAlpha * g));
  act[idx] = (_Float16)(g * sg * (lin + 1.0f));
}

// ---------------------------------------------------------------------------
// Host launcher
// ---------------------------------------------------------------------------
static inline size_t align256(size_t x) { return (x + 255) & ~(size_t)255; }

extern "C" void kernel_launch(void* const* d_in, const int* in_sizes, int n_in,
                              void* d_out, int out_size, void* d_ws, size_t ws_size,
                              hipStream_t stream) {
  const float* x    = (const float*)d_in[0];
  const float* ans  = (const float*)d_in[1];
  const float* wqkv = (const float*)d_in[2];
  const float* bqkv = (const float*)d_in[3];
  const float* wout = (const float*)d_in[4];
  const float* bout = (const float*)d_in[5];
  const float* mns  = (const float*)d_in[6];
  const float* w1   = (const float*)d_in[7];
  const float* b1   = (const float*)d_in[8];
  const float* w2   = (const float*)d_in[9];
  const float* b2   = (const float*)d_in[10];
  float* out = (float*)d_out;

  char* ws = (char*)d_ws;
  size_t off = 0;
  auto carve = [&](size_t bytes) { char* p = ws + off; off += align256(bytes); return p; };
  _Float16* t_h   = (_Float16*)carve((size_t)kT * kHid * 2);
  float*    qkv_f = (float*)   carve((size_t)kT * kQkvD * 4);
  _Float16* q_h   = (_Float16*)carve((size_t)kNH * kT * kHD * 2);
  _Float16* k_h   = (_Float16*)carve((size_t)kKV * kT * kHD * 2);
  _Float16* v_t   = (_Float16*)carve((size_t)kKV * kHD * kT * 2);
  _Float16* o_h   = (_Float16*)carve((size_t)kT * kNH * kHD * 2);
  float*    h_f   = (float*)   carve((size_t)kT * kHid * 4);
  _Float16* t2_h  = (_Float16*)carve((size_t)kT * kHid * 2);
  float*    u_f   = (float*)   carve((size_t)kT * 2 * kInter * 4);
  _Float16* act_h = (_Float16*)carve((size_t)kT * kInter * 2);
  (void)ws_size; (void)in_sizes; (void)n_in; (void)out_size;

  rmsnorm_h<<<kT, 256, 0, stream>>>(x, ans, t_h);

  gemm_h<<<dim3(kQkvD / kBN, kT / kBM), 256, 0, stream>>>(
      t_h, wqkv, bqkv, nullptr, qkv_f, kT, kQkvD, kHid);

  {
    const int total = kT * (kNH + 2 * kKV) * 32;
    rope_scatter<<<(total + 255) / 256, 256, 0, stream>>>(qkv_f, q_h, k_h, v_t);
  }

  attn_kernel<<<dim3(kNH, kT / 16), 32, 0, stream>>>(q_h, k_h, v_t, o_h);

  gemm_h<<<dim3(kHid / kBN, kT / kBM), 256, 0, stream>>>(
      o_h, wout, bout, x, h_f, kT, kHid, kNH * kHD);

  rmsnorm_h<<<kT, 256, 0, stream>>>(h_f, mns, t2_h);

  gemm_h<<<dim3((2 * kInter) / kBN, kT / kBM), 256, 0, stream>>>(
      t2_h, w1, b1, nullptr, u_f, kT, 2 * kInter, kHid);

  {
    const int total = kT * kInter;
    swiglu_act<<<(total + 255) / 256, 256, 0, stream>>>(u_f, act_h);
  }

  gemm_h<<<dim3(kHid / kBN, kT / kBM), 256, 0, stream>>>(
      act_h, w2, b2, h_f, out, kT, kHid, kInter);
}
